// Model6_36713380446240
// MI455X (gfx1250) — matrix operand order repaired and checked by reading, not compiled, not run on hardware
//
#include <hip/hip_runtime.h>

typedef __bf16 bf16_t;
typedef __attribute__((ext_vector_type(16))) __bf16 v16bf;
typedef __attribute__((ext_vector_type(8)))  __bf16 v8bf;
typedef __attribute__((ext_vector_type(2)))  __bf16 v2bf;
typedef __attribute__((ext_vector_type(8)))  float  v8f;
typedef __attribute__((ext_vector_type(8)))  float  v8fv;
typedef __attribute__((ext_vector_type(2)))  float  v2f;
typedef int v4i_t __attribute__((vector_size(16)));   // GCC-style int4 (builtin param type)

#define B_SZ 64
#define S_SZ 2048
#define IN_SZ 512
#define H_SZ 256
#define G_SZ 768                 // 3*H
#define M_SZ (S_SZ * B_SZ)       // 131072
#define LDS_PAD 16               // elements; keeps 16B chunk alignment, shifts banks per row

// ---------------- CDNA5 async global->LDS path (guarded; ds-store fallback) -------------
#if __has_builtin(__builtin_amdgcn_global_load_async_to_lds_b128)
#define HAVE_ASYNC_LDS 1
#else
#define HAVE_ASYNC_LDS 0
#endif

#if HAVE_ASYNC_LDS
static __device__ __forceinline__ void async_copy16(const void* g, void* l) {
    // generic LDS pointer low 32 bits == LDS byte offset (shared aperture); AS1 == flat addr
    __builtin_amdgcn_global_load_async_to_lds_b128(
        (__attribute__((address_space(1))) v4i_t*)(unsigned long long)g,
        (__attribute__((address_space(3))) v4i_t*)(unsigned)(unsigned long long)l,
        0, 0);
}
static __device__ __forceinline__ void wait_async_zero() {
#if __has_builtin(__builtin_amdgcn_s_wait_asynccnt)
    __builtin_amdgcn_s_wait_asynccnt(0);
#else
    asm volatile("s_wait_asynccnt 0" ::: "memory");
#endif
}
#endif

// ---------------- bf16 helpers (native converts -> v_cvt_pk_bf16_f32) ----------------
static __device__ __forceinline__ bf16_t f2bf(float f) { return (bf16_t)f; }
static __device__ __forceinline__ float  bf2f(bf16_t b) { return (float)b; }

static __device__ __forceinline__ float sigmoidf_fast(float x) {
    return 1.0f / (1.0f + __expf(-x));
}
static __device__ __forceinline__ float tanhf_fast(float x) {
    const float xc = fminf(fmaxf(x, -15.0f), 15.0f);
    const float e = __expf(2.0f * xc);
    return (e - 1.0f) / (e + 1.0f);
}

// Stage 64 weight rows of length KD (bf16) into LDS: smem[row*(KD+LDS_PAD) + k]
template<int KD>
static __device__ __forceinline__ void stage_panel(const bf16_t* __restrict__ wpanel,
                                                   bf16_t* smem) {
    const int chunksPerRow = KD / 8;                  // 16B chunks per row
    const int totalChunks  = 64 * chunksPerRow;       // divisible by 256
    const int strideE      = KD + LDS_PAD;
    for (int c = threadIdx.x; c < totalChunks; c += 256) {
        const int rowL = c / chunksPerRow;
        const int cc   = c - rowL * chunksPerRow;
        const bf16_t* g = wpanel + (size_t)rowL * KD + cc * 8;
        bf16_t*       l = smem + rowL * strideE + cc * 8;
#if HAVE_ASYNC_LDS
        async_copy16(g, l);
#else
        *(v8bf*)l = *(const v8bf*)g;
#endif
    }
#if HAVE_ASYNC_LDS
    wait_async_zero();
#endif
    __syncthreads();
}

// Read one 32x16 B fragment (col = l16, K = k0 + half*8 .. +7 and k0 + 16 + half*8 .. +7) from the staged panel
template<int KD>
static __device__ __forceinline__ v16bf b_frag(const bf16_t* smem, int nLocal, int k0, int half) {
    const bf16_t* bp = smem + nLocal * (KD + LDS_PAD) + k0 + half * 8;
    const v8bf b0 = *(const v8bf*)bp;
    const v8bf b1 = *(const v8bf*)(bp + 16);
    v16bf bf;
    #pragma unroll
    for (int i = 0; i < 8; ++i) { bf[i] = b0[i]; bf[8 + i] = b1[i]; }
    return bf;
}

// ---------- f32 -> bf16 weight conversion ----------
__global__ void cvt_kernel(const float* __restrict__ in, bf16_t* __restrict__ out, int n) {
    int i = blockIdx.x * blockDim.x + threadIdx.x;
    if (i < n) out[i] = f2bf(in[i]);
}

// ---------- fc1: h0(S,B,H) = x(B,S,IN) @ fc1_w(H,IN)^T + b, stored bf16 ----------
// Block = 256 threads = 8 waves. Block tile: 128(M) x 64(N). Wave: 16(M) x 64(N).
__global__ __launch_bounds__(256) void fc1_gemm(const float* __restrict__ x,
                                                const bf16_t* __restrict__ w,
                                                const float* __restrict__ bias,
                                                bf16_t* __restrict__ out) {
    __shared__ bf16_t sB[64 * (IN_SZ + LDS_PAD)];     // 66 KB
    const int lane = threadIdx.x & 31;
    const int wave = threadIdx.x >> 5;
    const int half = lane >> 4;
    const int l16  = lane & 15;
    const int mBase = blockIdx.x * 128 + wave * 16;
    const int nBase = blockIdx.y * 64;

    stage_panel<IN_SZ>(w + (size_t)nBase * IN_SZ, sB);

    // row m = s*B + b ; x is (B,S,IN)
    const int row = mBase + l16;
    const int s = row >> 6;           // / B_SZ
    const int b = row & 63;           // % B_SZ
    const float* __restrict__ arow = x + (size_t)b * (S_SZ * IN_SZ) + (size_t)s * IN_SZ;

    v8f acc[4];
    #pragma unroll
    for (int t = 0; t < 4; ++t)
        #pragma unroll
        for (int i = 0; i < 8; ++i) acc[t][i] = 0.0f;

    for (int k0 = 0; k0 < IN_SZ; k0 += 32) {
        // A fragment (16x32 bf16): lane half picks K sub-ranges per ISA layout
        v16bf a;
        const v8fv f1 = *(const v8fv*)(arow + k0 + half * 8);
        const v8fv f2 = *(const v8fv*)(arow + k0 + 16 + half * 8);
        #pragma unroll
        for (int i = 0; i < 8; ++i) { a[i] = f2bf(f1[i]); a[8 + i] = f2bf(f2[i]); }

        #pragma unroll
        for (int t = 0; t < 4; ++t) {
            const v16bf bf = b_frag<IN_SZ>(sB, t * 16 + l16, k0, half);
            acc[t] = __builtin_amdgcn_wmma_f32_16x16x32_bf16(
                false, a, false, bf, (short)0, acc[t], false, false);
        }
    }

    #pragma unroll
    for (int t = 0; t < 4; ++t) {
        const int n = nBase + t * 16 + l16;
        const float bv = bias[n];
        #pragma unroll
        for (int r = 0; r < 8; ++r) {
            const int m = mBase + half * 8 + r;
            out[(size_t)m * H_SZ + n] = f2bf(acc[t][r] + bv);
        }
    }
}

// ---------- qrnn layer GEMM: y(S,B,3H) = h(S,B,H) @ w(3H,H)^T + b, fused act ----------
__global__ __launch_bounds__(256) void qrnn_gemm(const bf16_t* __restrict__ hin,
                                                 const bf16_t* __restrict__ w,
                                                 const float* __restrict__ bias,
                                                 bf16_t* __restrict__ y) {
    __shared__ bf16_t sB[64 * (H_SZ + LDS_PAD)];      // 34 KB
    const int lane = threadIdx.x & 31;
    const int wave = threadIdx.x >> 5;
    const int half = lane >> 4;
    const int l16  = lane & 15;
    const int mBase = blockIdx.x * 128 + wave * 16;
    const int nBase = blockIdx.y * 64;

    stage_panel<H_SZ>(w + (size_t)nBase * H_SZ, sB);

    const bf16_t* __restrict__ arow = hin + (size_t)(mBase + l16) * H_SZ;

    v8f acc[4];
    #pragma unroll
    for (int t = 0; t < 4; ++t)
        #pragma unroll
        for (int i = 0; i < 8; ++i) acc[t][i] = 0.0f;

    #pragma unroll
    for (int k0 = 0; k0 < H_SZ; k0 += 32) {
        v16bf a;
        const v8bf lo = *(const v8bf*)(arow + k0 + half * 8);
        const v8bf hi = *(const v8bf*)(arow + k0 + 16 + half * 8);
        #pragma unroll
        for (int i = 0; i < 8; ++i) { a[i] = lo[i]; a[8 + i] = hi[i]; }

        #pragma unroll
        for (int t = 0; t < 4; ++t) {
            const v16bf bf = b_frag<H_SZ>(sB, t * 16 + l16, k0, half);
            acc[t] = __builtin_amdgcn_wmma_f32_16x16x32_bf16(
                false, a, false, bf, (short)0, acc[t], false, false);
        }
    }

    #pragma unroll
    for (int t = 0; t < 4; ++t) {
        const int n = nBase + t * 16 + l16;
        const float bv = bias[n];
        const bool is_z = (n < H_SZ);   // uniform within a 16-wide tile (64 | 256)
        #pragma unroll
        for (int r = 0; r < 8; ++r) {
            const int m = mBase + half * 8 + r;
            float v = acc[t][r] + bv;
            v = is_z ? tanhf_fast(v) : sigmoidf_fast(v);
            y[(size_t)m * G_SZ + n] = f2bf(v);
        }
    }
}

// ---------- sequential scan over S; one thread per (b, pair-of-h) ----------
__global__ __launch_bounds__(256) void qrnn_scan(const bf16_t* __restrict__ y,
                                                 bf16_t* __restrict__ hout_bf,
                                                 float* __restrict__ hout_f32,
                                                 float* __restrict__ c_last) {
    const int tid = blockIdx.x * blockDim.x + threadIdx.x;   // 0..8191
    const int b  = tid >> 7;              // batch
    const int hh = (tid & 127) * 2;       // pair of h-channels
    size_t base = (size_t)b * G_SZ + hh;  // element index into y[s=0]
    const size_t strideS = (size_t)B_SZ * G_SZ;
    float c0 = 0.0f, c1 = 0.0f;
    for (int s = 0; s < S_SZ; ++s) {
        __builtin_prefetch(y + base + 8 * strideS);          // speculative, safe past end
        const v2bf zv = *(const v2bf*)(y + base);
        const v2bf fv = *(const v2bf*)(y + base + H_SZ);
        const v2bf ov = *(const v2bf*)(y + base + 2 * H_SZ);
        const float f0 = bf2f(fv[0]), f1v = bf2f(fv[1]);
        c0 = f0 * c0 + (1.0f - f0) * bf2f(zv[0]);
        c1 = f1v * c1 + (1.0f - f1v) * bf2f(zv[1]);
        const float h0 = bf2f(ov[0]) * c0;
        const float h1 = bf2f(ov[1]) * c1;
        const size_t oidx = (size_t)s * (B_SZ * H_SZ) + b * H_SZ + hh;
        if (hout_bf) {
            v2bf hb; hb[0] = f2bf(h0); hb[1] = f2bf(h1);
            *(v2bf*)(hout_bf + oidx) = hb;
        }
        if (hout_f32) {
            v2f hf; hf[0] = h0; hf[1] = h1;
            *(v2f*)(hout_f32 + oidx) = hf;
        }
        base += strideS;
    }
    c_last[b * H_SZ + hh]     = c0;
    c_last[b * H_SZ + hh + 1] = c1;
}

// ---------- head: logits + softmax ----------
__global__ void head_kernel(const float* __restrict__ hlast,
                            const float* __restrict__ fc_w,
                            const float* __restrict__ fc_b,
                            float* __restrict__ out) {
    const int b = threadIdx.x;
    if (b >= B_SZ) return;
    float logits[10];
    float mx = -1e30f;
    const float* __restrict__ hr = hlast + b * H_SZ;
    for (int j = 0; j < 10; ++j) {
        float acc = fc_b[j];
        const float* __restrict__ wr = fc_w + j * H_SZ;
        for (int k = 0; k < H_SZ; ++k) acc += hr[k] * wr[k];
        logits[j] = acc;
        mx = fmaxf(mx, acc);
    }
    float sum = 0.0f;
    for (int j = 0; j < 10; ++j) { logits[j] = __expf(logits[j] - mx); sum += logits[j]; }
    const float inv = 1.0f / sum;
    for (int j = 0; j < 10; ++j) out[b * 10 + j] = logits[j] * inv;
}

extern "C" void kernel_launch(void* const* d_in, const int* in_sizes, int n_in,
                              void* d_out, int out_size, void* d_ws, size_t ws_size,
                              hipStream_t stream) {
    const float* x      = (const float*)d_in[0];
    const float* fc1_w  = (const float*)d_in[1];
    const float* fc1_b  = (const float*)d_in[2];
    const float* qrnn_w = (const float*)d_in[3];
    const float* qrnn_b = (const float*)d_in[4];
    const float* fc_w   = (const float*)d_in[5];
    const float* fc_b   = (const float*)d_in[6];
    float* out = (float*)d_out;

    // workspace layout (bytes):
    //   [0, 256K)      fc1_w bf16   (131072 elems)
    //   [256K, ~1.8M)  qrnn_w bf16  (786432 elems)
    //   [+64MB)        h buffer bf16 (S*B*H)
    //   [+192MB)       y buffer bf16 (S*B*3H)
    char* ws = (char*)d_ws;
    bf16_t* w_fc1  = (bf16_t*)ws;
    bf16_t* w_qrnn = (bf16_t*)(ws + 262144);
    bf16_t* hbuf   = (bf16_t*)(ws + 262144 + 1572864);
    bf16_t* ybuf   = (bf16_t*)(ws + 262144 + 1572864 + (size_t)S_SZ * B_SZ * H_SZ * 2);

    cvt_kernel<<<(131072 + 255) / 256, 256, 0, stream>>>(fc1_w, w_fc1, 131072);
    cvt_kernel<<<(786432 + 255) / 256, 256, 0, stream>>>(qrnn_w, w_qrnn, 786432);

    dim3 g1(M_SZ / 128, H_SZ / 64);   // (1024, 4)
    fc1_gemm<<<g1, 256, 0, stream>>>(x, w_fc1, fc1_b, hbuf);

    float* h_out_f32 = out;                                      // (S,B,H) f32
    float* cls       = out + (size_t)S_SZ * B_SZ * H_SZ;         // (B,10)
    float* hn        = cls + B_SZ * 10;                          // (4,B,H)

    dim3 g2(M_SZ / 128, G_SZ / 64);   // (1024, 12)
    for (int l = 0; l < 4; ++l) {
        qrnn_gemm<<<g2, 256, 0, stream>>>(hbuf,
                                          w_qrnn + (size_t)l * G_SZ * H_SZ,
                                          qrnn_b + l * G_SZ,
                                          ybuf);
        const bool last = (l == 3);
        qrnn_scan<<<(B_SZ * H_SZ / 2) / 256, 256, 0, stream>>>(
            ybuf,
            last ? (bf16_t*)nullptr : hbuf,
            last ? h_out_f32 : (float*)nullptr,
            hn + (size_t)l * B_SZ * H_SZ);
    }

    head_kernel<<<1, 64, 0, stream>>>(h_out_f32 + (size_t)(S_SZ - 1) * B_SZ * H_SZ,
                                      fc_w, fc_b, cls);
}
